// ReadingGCNStage_28063316312877
// MI455X (gfx1250) — matrix operand order repaired and checked by reading, not compiled, not run on hardware
//
#include <hip/hip_runtime.h>

typedef _Float16 f16;
typedef __attribute__((ext_vector_type(16))) _Float16 v16h;
typedef __attribute__((ext_vector_type(8)))  _Float16 v8h;
typedef __attribute__((ext_vector_type(8)))  float    v8f;

#define N_R    20000
#define N_S    5000
#define DIM    512
#define NLAYER 3

#define BM  128
#define BN  128
#define BK  32
#define LDT 40   // padded f16 elements per LDS tile row (40*2 = 80B, keeps 16B alignment)

// ---------------------------------------------------------------------------
// Prep kernels
// ---------------------------------------------------------------------------

// One block per reading-row: convert A row f32->f16 and produce 1/(rowsum+eps).
__global__ __launch_bounds__(256) void k_rowcvt(const float* __restrict__ A,
                                                f16* __restrict__ A16,
                                                float* __restrict__ invdeg,
                                                int ns) {
  __shared__ float red[256];
  const int row = blockIdx.x;
  const float* arow = A + (size_t)row * ns;
  f16* orow = A16 + (size_t)row * ns;
  float s = 0.f;
  for (int j = threadIdx.x; j < ns; j += 256) {
    float v = arow[j];
    orow[j] = (f16)v;
    s += v;
  }
  red[threadIdx.x] = s;
  __syncthreads();
  for (int off = 128; off > 0; off >>= 1) {
    if ((int)threadIdx.x < off) red[threadIdx.x] += red[threadIdx.x + off];
    __syncthreads();
  }
  if (threadIdx.x == 0) invdeg[row] = 1.0f / (red[0] + 1e-6f);
}

// Deterministic column sums: stage 1 partials over row chunks (coalesced).
__global__ __launch_bounds__(256) void k_colpart(const float* __restrict__ A,
                                                 float* __restrict__ part,
                                                 int nr, int ns, int rows_per) {
  int j = blockIdx.x * 256 + threadIdx.x;
  int r0 = blockIdx.y * rows_per;
  int r1 = r0 + rows_per; if (r1 > nr) r1 = nr;
  if (j < ns) {
    float s = 0.f;
    for (int r = r0; r < r1; ++r) s += A[(size_t)r * ns + j];
    part[(size_t)blockIdx.y * ns + j] = s;
  }
}

__global__ __launch_bounds__(256) void k_colfin(const float* __restrict__ part,
                                                float* __restrict__ invdeg,
                                                int ns, int nchunks) {
  int j = blockIdx.x * 256 + threadIdx.x;
  if (j < ns) {
    float s = 0.f;
    for (int c = 0; c < nchunks; ++c) s += part[(size_t)c * ns + j];
    invdeg[j] = 1.0f / (s + 1e-6f);
  }
}

__global__ __launch_bounds__(256) void k_cvt(const float* __restrict__ src,
                                             f16* __restrict__ dst, long n) {
  long i = (long)blockIdx.x * blockDim.x + threadIdx.x;
  long stride = (long)gridDim.x * blockDim.x;
  for (; i < n; i += stride) dst[i] = (f16)src[i];
}

// pack two f16 into one dword for b32 LDS stores
__device__ __forceinline__ unsigned pack2(_Float16 a, _Float16 b) {
  union { _Float16 h[2]; unsigned u; } c;
  c.h[0] = a; c.h[1] = b;
  return c.u;
}

// ---------------------------------------------------------------------------
// Tiled WMMA GEMM:  C[M,N] = L[M,K] * R[K,N]
//   AT=false: L stored row-major [M][lda]
//   AT=true : L stored transposed as S[K][lda] with L[m][k] = S[k][m]
//   R always stored row-major [K][ldb]; transposed into LDS as [N][K].
// EPI: 0 -> C16 = f16(acc * rowscale[m])
//      1 -> Cf  = acc
//      2 -> v = relu(acc + Cin); C16 = f16(v); if (Cf) Cf = v
// ---------------------------------------------------------------------------
template <bool AT, int EPI>
__global__ __launch_bounds__(256) void gemm_wmma(const f16* __restrict__ Aop, int lda,
                                                 const f16* __restrict__ Bop, int ldb,
                                                 float* __restrict__ Cf,
                                                 f16* __restrict__ C16,
                                                 const float* __restrict__ rowscale,
                                                 const float* __restrict__ Cin,
                                                 int M, int N, int K) {
  __shared__ alignas(16) _Float16 sA[BM * LDT];
  __shared__ alignas(16) _Float16 sB[BN * LDT];

  const int tid  = threadIdx.x;
  const int lane = tid & 31;
  const int wave = tid >> 5;
  const int wm   = wave >> 1;   // 0..3 -> 32-row slice
  const int wn   = wave & 1;    // 0..1 -> 64-col slice
  const int m0 = blockIdx.y * BM;
  const int n0 = blockIdx.x * BN;
  const int halflane = lane & 15;
  const int hisel    = lane >> 4;   // 0 or 1

  const bool interior_m = (m0 + BM <= M);

  v8f acc[2][4];
#pragma unroll
  for (int i = 0; i < 2; ++i)
#pragma unroll
    for (int j = 0; j < 4; ++j) {
      v8f z = {0.f, 0.f, 0.f, 0.f, 0.f, 0.f, 0.f, 0.f};
      acc[i][j] = z;
    }

  for (int kk = 0; kk < K; kk += BK) {
    const bool fullk = (kk + BK <= K);

    // ---- stage L tile into sA[m][k] ----
    if (!AT) {
      const int c = tid & 3;        // 8-wide k chunk
      const int r = tid >> 2;       // 0..63
      if (interior_m && fullk) {
        // CDNA5 async direct-to-LDS path (ASYNCcnt), no VGPR round-trip.
#pragma unroll
        for (int h = 0; h < 2; ++h) {
          const int rr = r + h * 64;
          const f16* gp = Aop + (size_t)(m0 + rr) * lda + kk + c * 8;
          const unsigned ldsoff =
              (unsigned)(uintptr_t)(sA + rr * LDT + c * 8);
          asm volatile("global_load_async_to_lds_b128 %0, %1, off"
                       :: "v"(ldsoff), "v"(gp) : "memory");
        }
        asm volatile("s_wait_asynccnt 0" ::: "memory");
      } else {
#pragma unroll
        for (int h = 0; h < 2; ++h) {
          const int rr = r + h * 64;
          const int gm = m0 + rr;
          const int gk = kk + c * 8;
          v8h val;
#pragma unroll
          for (int e = 0; e < 8; ++e) val[e] = (_Float16)0.f;
          if (gm < M && gk < K)
            val = *reinterpret_cast<const v8h*>(Aop + (size_t)gm * lda + gk);
          *reinterpret_cast<v8h*>(sA + rr * LDT + c * 8) = val;
        }
      }
    } else {
      // transposed-storage left operand: coalesced global rows, packed b32
      // scatter into sA[m][k] (adjacent k pairs share one dword).
      const int c  = tid & 15;      // 8-wide m chunk
      const int kp = tid >> 4;      // 0..15 -> k rows 2kp, 2kp+1
      const int kj = kp * 2;
      const int gm = m0 + c * 8;
      v8h v0, v1;
#pragma unroll
      for (int e = 0; e < 8; ++e) { v0[e] = (_Float16)0.f; v1[e] = (_Float16)0.f; }
      if (gm < M) {
        const int gk0 = kk + kj;
        if (gk0 < K)
          v0 = *reinterpret_cast<const v8h*>(Aop + (size_t)gk0 * lda + gm);
        if (gk0 + 1 < K)
          v1 = *reinterpret_cast<const v8h*>(Aop + (size_t)(gk0 + 1) * lda + gm);
      }
#pragma unroll
      for (int e = 0; e < 8; ++e)
        *reinterpret_cast<unsigned*>(sA + (c * 8 + e) * LDT + kj) =
            pack2(v0[e], v1[e]);
    }

    // ---- stage R tile transposed into sB[n][k], packed b32 stores ----
    {
      const int c  = tid & 15;      // 8-wide n chunk
      const int kp = tid >> 4;      // 0..15 -> k rows 2kp, 2kp+1
      const int kj = kp * 2;
      const int gn = n0 + c * 8;
      const int gk0 = kk + kj;
      v8h v0, v1;
#pragma unroll
      for (int e = 0; e < 8; ++e) { v0[e] = (_Float16)0.f; v1[e] = (_Float16)0.f; }
      if (gk0 < K)
        v0 = *reinterpret_cast<const v8h*>(Bop + (size_t)gk0 * ldb + gn);
      if (gk0 + 1 < K)
        v1 = *reinterpret_cast<const v8h*>(Bop + (size_t)(gk0 + 1) * ldb + gn);
#pragma unroll
      for (int e = 0; e < 8; ++e)
        *reinterpret_cast<unsigned*>(sB + (c * 8 + e) * LDT + kj) =
            pack2(v0[e], v1[e]);
    }
    __syncthreads();

    // ---- fragment gathers (wave32 WMMA layouts) ----
    // A fragment 16x32: lane holds row (lane&15); K chunks {hisel*8 .. +7} and {+16..+23}
    v16h af[2];
#pragma unroll
    for (int i = 0; i < 2; ++i) {
      const int row = wm * 32 + i * 16 + halflane;
      const _Float16* p = sA + row * LDT + hisel * 8;
      v8h lo = *reinterpret_cast<const v8h*>(p);
      v8h hi = *reinterpret_cast<const v8h*>(p + 16);
#pragma unroll
      for (int e = 0; e < 8; ++e) { af[i][e] = lo[e]; af[i][e + 8] = hi[e]; }
    }
    // B fragment 32x16: lane holds col (lane&15); K chunks {hisel*8 .. +7} and {+16..+23}
    v16h bf[4];
#pragma unroll
    for (int j = 0; j < 4; ++j) {
      const int row = wn * 64 + j * 16 + halflane;
      const _Float16* p = sB + row * LDT + hisel * 8;
      v8h lo = *reinterpret_cast<const v8h*>(p);
      v8h hi = *reinterpret_cast<const v8h*>(p + 16);
#pragma unroll
      for (int e = 0; e < 8; ++e) { bf[j][e] = lo[e]; bf[j][e + 8] = hi[e]; }
    }

#pragma unroll
    for (int i = 0; i < 2; ++i)
#pragma unroll
      for (int j = 0; j < 4; ++j)
        acc[i][j] = __builtin_amdgcn_wmma_f32_16x16x32_f16(
            false, af[i], false, bf[j], (short)0, acc[i][j], false, false);

    __syncthreads();
  }

  // ---- epilogue: C/D layout -> VGPR v holds M = v + 8*hisel, N = lane&15 ----
#pragma unroll
  for (int i = 0; i < 2; ++i) {
#pragma unroll
    for (int j = 0; j < 4; ++j) {
      const int mb = m0 + wm * 32 + i * 16 + hisel * 8;
      const int n  = n0 + wn * 64 + j * 16 + halflane;
#pragma unroll
      for (int v = 0; v < 8; ++v) {
        const int gm = mb + v;
        if (gm < M) {
          float x = acc[i][j][v];
          const size_t idx = (size_t)gm * N + n;
          if (EPI == 0) {
            x *= rowscale[gm];
            C16[idx] = (f16)x;
          } else if (EPI == 1) {
            Cf[idx] = x;
          } else {
            x += Cin[idx];
            x = fmaxf(x, 0.0f);
            C16[idx] = (f16)x;
            if (Cf) Cf[idx] = x;
          }
        }
      }
    }
  }
}

// ---------------------------------------------------------------------------
// Host launcher
// ---------------------------------------------------------------------------
extern "C" void kernel_launch(void* const* d_in, const int* in_sizes, int n_in,
                              void* d_out, int out_size, void* d_ws, size_t ws_size,
                              hipStream_t stream) {
  const float* h_s = (const float*)d_in[0];
  const float* A   = (const float*)d_in[1];
  const float* r_e = (const float*)d_in[2];
  const float* Wr  = (const float*)d_in[3];
  const float* Ws  = (const float*)d_in[4];
  const float* Vr  = (const float*)d_in[5];
  const float* Vs  = (const float*)d_in[6];
  float* out = (float*)d_out;

  char* p = (char*)d_ws;
  auto alloc = [&](size_t bytes) -> void* {
    void* r = (void*)p;
    p += (bytes + 255) & ~(size_t)255;
    return r;
  };

  const size_t DD = (size_t)DIM * DIM;
  f16*   A16      = (f16*)alloc((size_t)N_R * N_S * sizeof(f16));
  float* invdeg_r = (float*)alloc((size_t)N_R * sizeof(float));
  float* invdeg_s = (float*)alloc((size_t)N_S * sizeof(float));
  float* colpart  = (float*)alloc((size_t)80 * N_S * sizeof(float));
  f16*   hs16a    = (f16*)alloc((size_t)N_S * DIM * sizeof(f16));
  f16*   hs16b    = (f16*)alloc((size_t)N_S * DIM * sizeof(f16));
  f16*   hr16a    = (f16*)alloc((size_t)N_R * DIM * sizeof(f16));
  f16*   hr16b    = (f16*)alloc((size_t)N_R * DIM * sizeof(f16));
  f16*   mr16     = (f16*)alloc((size_t)N_R * DIM * sizeof(f16));
  f16*   ms16     = (f16*)alloc((size_t)N_S * DIM * sizeof(f16));
  float* tmp_r    = (float*)alloc((size_t)N_R * DIM * sizeof(float));
  float* tmp_s    = (float*)alloc((size_t)N_S * DIM * sizeof(float));
  f16*   Wr16     = (f16*)alloc(NLAYER * DD * sizeof(f16));
  f16*   Ws16     = (f16*)alloc(NLAYER * DD * sizeof(f16));
  f16*   Vr16     = (f16*)alloc(NLAYER * DD * sizeof(f16));
  f16*   Vs16     = (f16*)alloc(NLAYER * DD * sizeof(f16));

  // --- preprocessing: degrees + f16 conversions ---
  k_rowcvt<<<N_R, 256, 0, stream>>>(A, A16, invdeg_r, N_S);
  k_colpart<<<dim3((N_S + 255) / 256, 80), 256, 0, stream>>>(A, colpart, N_R, N_S, 250);
  k_colfin<<<(N_S + 255) / 256, 256, 0, stream>>>(colpart, invdeg_s, N_S, 80);

  auto cvt = [&](const float* s, f16* d, long n) {
    long blocks = (n + 255) / 256;
    if (blocks > 16384) blocks = 16384;
    k_cvt<<<(int)blocks, 256, 0, stream>>>(s, d, n);
  };
  cvt(h_s, hs16a, (long)N_S * DIM);
  cvt(r_e, hr16a, (long)N_R * DIM);
  cvt(Wr, Wr16, (long)NLAYER * DD);
  cvt(Ws, Ws16, (long)NLAYER * DD);
  cvt(Vr, Vr16, (long)NLAYER * DD);
  cvt(Vs, Vs16, (long)NLAYER * DD);

  const dim3 blk(256);
  const dim3 gR(DIM / BN, (N_R + BM - 1) / BM);  // (4, 157)
  const dim3 gS(DIM / BN, (N_S + BM - 1) / BM);  // (4, 40)

  f16 *hs_cur = hs16a, *hs_nxt = hs16b;
  f16 *hr_cur = hr16a, *hr_nxt = hr16b;

  for (int l = 0; l < NLAYER; ++l) {
    const bool last = (l == NLAYER - 1);

    // m_r16 = f16( (A @ h_s) / deg_r )
    gemm_wmma<false, 0><<<gR, blk, 0, stream>>>(
        A16, N_S, hs_cur, DIM, nullptr, mr16, invdeg_r, nullptr, N_R, DIM, N_S);

    if (!last) {
      // m_s16 = f16( (A^T @ h_r) / deg_s )   (left operand stored transposed)
      gemm_wmma<true, 0><<<gS, blk, 0, stream>>>(
          A16, N_S, hr_cur, DIM, nullptr, ms16, invdeg_s, nullptr, N_S, DIM, N_R);
    }

    // h_r_new = relu(m_r @ Ws[l] + h_r @ Wr[l])
    gemm_wmma<false, 1><<<gR, blk, 0, stream>>>(
        mr16, DIM, Ws16 + (size_t)l * DD, DIM, tmp_r, nullptr, nullptr, nullptr,
        N_R, DIM, DIM);
    gemm_wmma<false, 2><<<gR, blk, 0, stream>>>(
        hr_cur, DIM, Wr16 + (size_t)l * DD, DIM, last ? out : nullptr, hr_nxt,
        nullptr, tmp_r, N_R, DIM, DIM);

    if (!last) {
      // h_s_new = relu(m_s @ Vr[l] + h_s @ Vs[l])
      gemm_wmma<false, 1><<<gS, blk, 0, stream>>>(
          ms16, DIM, Vr16 + (size_t)l * DD, DIM, tmp_s, nullptr, nullptr, nullptr,
          N_S, DIM, DIM);
      gemm_wmma<false, 2><<<gS, blk, 0, stream>>>(
          hs_cur, DIM, Vs16 + (size_t)l * DD, DIM, nullptr, hs_nxt,
          nullptr, tmp_s, N_S, DIM, DIM);
    }

    f16* t;
    t = hr_cur; hr_cur = hr_nxt; hr_nxt = t;
    t = hs_cur; hs_cur = hs_nxt; hs_nxt = t;
  }
}
